// SparseUNetRPN_38079180046735
// MI455X (gfx1250) — compile-verified, matrix operand order checked by reading, not run on hardware
//
#include <hip/hip_runtime.h>
#include <math.h>

// ---------------------------------------------------------------------------
// CDNA5 (gfx1250) wave32 WMMA types
// ---------------------------------------------------------------------------
typedef __attribute__((ext_vector_type(16))) __bf16 v16bf;
typedef __attribute__((ext_vector_type(8)))  float  v8f;

union Frag16 { v16bf v; unsigned int u[8]; };

#define FLAG_BN   1
#define FLAG_RELU 2
#define FLAG_RES  4

// ---------------------------------------------------------------------------
// Per-channel batchnorm statistics (training mode: mean / rsqrt(var+eps))
// ---------------------------------------------------------------------------
__global__ __launch_bounds__(256) void k_bnstats(const float* __restrict__ x, int S,
                                                 float* __restrict__ mean,
                                                 float* __restrict__ inv) {
  const int c = blockIdx.x;
  const float* p = x + (size_t)c * S;
  float s = 0.f, s2 = 0.f;
  for (int i = threadIdx.x; i < S; i += blockDim.x) {
    float v = p[i];
    s += v; s2 += v * v;
  }
  __shared__ float sh[256], sh2[256];
  sh[threadIdx.x] = s; sh2[threadIdx.x] = s2;
  __syncthreads();
  for (int o = 128; o > 0; o >>= 1) {
    if (threadIdx.x < o) {
      sh[threadIdx.x]  += sh[threadIdx.x + o];
      sh2[threadIdx.x] += sh2[threadIdx.x + o];
    }
    __syncthreads();
  }
  if (threadIdx.x == 0) {
    float m = sh[0] / (float)S;
    float v = sh2[0] / (float)S - m * m;
    mean[c] = m;
    inv[c]  = rsqrtf(v + 1e-5f);
  }
}

// ---------------------------------------------------------------------------
// Fused BN-apply / residual-add / ReLU elementwise kernel
// ---------------------------------------------------------------------------
__global__ __launch_bounds__(256) void k_affine(const float* __restrict__ x,
                                                const float* __restrict__ res,
                                                float* __restrict__ y,
                                                const float* __restrict__ mean,
                                                const float* __restrict__ inv,
                                                const float* __restrict__ g,
                                                const float* __restrict__ b,
                                                int C, int S, int flags) {
  long tot = (long)C * S;
  long i = (long)blockIdx.x * blockDim.x + threadIdx.x;
  if (i >= tot) return;
  int c = (int)(i / S);
  float v = x[i];
  if (flags & FLAG_BN)   v = (v - mean[c]) * inv[c] * g[c] + b[c];
  if (flags & FLAG_RES)  v += res[i];
  if (flags & FLAG_RELU) v = fmaxf(v, 0.f);
  y[i] = v;
}

// ---------------------------------------------------------------------------
// Implicit-GEMM 3D conv via V_WMMA_F32_16X16X32_BF16, f32 accumulate.
// Geometry templated (k/stride/pad are compile-time: 3/1/1, 2/2/0, 1/1/0) so
// all im2col divisions strength-reduce. One wave = 16(co) x 32(voxel) tile:
// one A panel feeds two back-to-back WMMAs (acc0/acc1). Voxel coordinates are
// decoded once per tile into a per-wave LDS table, not per K-chunk.
// Dual input pointer reads a channel-concat without materializing it.
// ---------------------------------------------------------------------------
template <int KK, int ST, int PD>
__global__ __launch_bounds__(256) void k_conv_wmma(
    const float* __restrict__ xA, int CiA, const float* __restrict__ xB,
    const float* __restrict__ w, float* __restrict__ y,
    int Ci, int Co,
    int inD, int inH, int inW,
    int outD, int outH, int outW) {
  constexpr int K2 = KK * KK, K3 = K2 * KK;
  const int K = Ci * K3;
  const int inHW = inH * inW, inS = inD * inHW;
  const int outHW = outH * outW, outS = outD * outHW;

  const int wid  = threadIdx.x >> 5;
  const int lane = threadIdx.x & 31;
  const int numNT = (outS + 31) >> 5;
  const bool valid = (blockIdx.x * 8 + wid) < numNT;
  int nTile = blockIdx.x * 8 + wid;
  if (nTile >= numNT) nTile = numNT - 1;       // clamp so all waves hit barriers
  const int nBase  = nTile << 5;
  const int coBase = blockIdx.y << 4;

  __shared__ __bf16 lA[8][16 * 32];            // weight panel (M=16 x K=32)
  __shared__ __bf16 lB[8][32 * 32];            // input patch  (K=32 x N=32)
  __shared__ int    lvD[8][32], lvH[8][32], lvW[8][32];
  __bf16* sA = lA[wid];
  __bf16* sB = lB[wid];
  int* vD = lvD[wid]; int* vH = lvH[wid]; int* vW = lvW[wid];

  // ---- decode this wave's 32 output voxels once (one voxel per lane)
  {
    int voxel = nBase + lane;
    int idb = -(1 << 20), ihb = idb, iwb = idb;  // forces OOB -> zero fill
    if (voxel < outS) {
      int od = voxel / outHW; int r2 = voxel - od * outHW;
      int oh = r2 / outW;     int ow = r2 - oh * outW;
      idb = od * ST - PD; ihb = oh * ST - PD; iwb = ow * ST - PD;
    }
    vD[lane] = idb; vH[lane] = ihb; vW[lane] = iwb;
  }
  __syncthreads();

  v8f acc0 = {}, acc1 = {};
  const int n     = lane & 15;
  const int khalf = (lane & 16) ? 8 : 0;

  for (int kb = 0; kb < K; kb += 32) {
    // ---- stage A: weights, layout w[co][ci*k^3] is already GEMM-contiguous
    for (int t = lane; t < 512; t += 32) {
      int m = t >> 5, kk = t & 31;
      int co = coBase + m, fk = kb + kk;
      float v = 0.f;
      if (co < Co && fk < K) v = w[(size_t)co * K + fk];
      sA[t] = (__bf16)v;
    }
    // ---- stage B: gathered input patch (im2col on the fly, div-free inner)
    for (int t = lane; t < 1024; t += 32) {
      int kk = t >> 5, nn = t & 31;
      int fk = kb + kk;
      float v = 0.f;
      if (fk < K) {
        int ci = fk / K3; int r = fk - ci * K3;          // constant divisors
        int kd = r / K2;  r -= kd * K2;
        int kh = r / KK;  int kw = r - kh * KK;
        int id = vD[nn] + kd, ih = vH[nn] + kh, iw = vW[nn] + kw;
        if ((unsigned)id < (unsigned)inD && (unsigned)ih < (unsigned)inH &&
            (unsigned)iw < (unsigned)inW) {
          const float* src = (ci < CiA) ? (xA + (size_t)ci * inS)
                                        : (xB + (size_t)(ci - CiA) * inS);
          v = src[id * inHW + ih * inW + iw];
        }
      }
      sB[t] = (__bf16)v;
    }
    __syncthreads();

    // ---- build 16-bit fragments per ISA §7.12.2 layout; two WMMAs share A
    Frag16 a, b0, b1;
    const unsigned short* sA16 = (const unsigned short*)sA;
    const unsigned short* sB16 = (const unsigned short*)sB;
    const int m = lane & 15;
#pragma unroll
    for (int j = 0; j < 8; ++j) {
      int k0 = (j < 4 ? 2 * j : 16 + 2 * (j - 4)) + khalf;
      a.u[j] = *(const unsigned int*)(sA16 + m * 32 + k0);   // packed pair
      unsigned lo0 = sB16[k0 * 32 + n];
      unsigned hi0 = sB16[(k0 + 1) * 32 + n];
      b0.u[j] = lo0 | (hi0 << 16);
      unsigned lo1 = sB16[k0 * 32 + 16 + n];
      unsigned hi1 = sB16[(k0 + 1) * 32 + 16 + n];
      b1.u[j] = lo1 | (hi1 << 16);
    }
    acc0 = __builtin_amdgcn_wmma_f32_16x16x32_bf16(
        false, a.v, false, b0.v, (short)0, acc0, false, false);
    acc1 = __builtin_amdgcn_wmma_f32_16x16x32_bf16(
        false, a.v, false, b1.v, (short)0, acc1, false, false);
    __syncthreads();
  }

  // ---- store C/D fragments: VGPR r -> M=r (lanes 0-15) / M=r+8 (lanes 16-31)
  const int rowOff = (lane & 16) ? 8 : 0;
  const int vox0 = nBase + n;
  const int vox1 = nBase + 16 + n;
#pragma unroll
  for (int r = 0; r < 8; ++r) {
    int co = coBase + r + rowOff;
    if (valid && co < Co) {
      if (vox0 < outS) y[(size_t)co * outS + vox0] = acc0[r];
      if (vox1 < outS) y[(size_t)co * outS + vox1] = acc1[r];
    }
  }
}

// ---------------------------------------------------------------------------
// Transposed conv k=2 s=2 (exact 2x upsample): per output-parity it is a
// 1x1 conv with K = Ci -> same WMMA tiling, grid.z enumerates the 8 parities.
// Weight layout (Ci, Co, 2,2,2).
// ---------------------------------------------------------------------------
__global__ __launch_bounds__(256) void k_tconv_wmma(
    const float* __restrict__ x, const float* __restrict__ w,
    float* __restrict__ y, int Ci, int Co, int inD, int inH, int inW) {
  const int inHW = inH * inW, inS = inD * inHW;
  const int outH = 2 * inH, outW = 2 * inW;
  const int outHW = outH * outW, outS = 2 * inD * outHW;
  const int par = blockIdx.z;
  const int pd = (par >> 2) & 1, ph = (par >> 1) & 1, pw = par & 1;
  const int widx = pd * 4 + ph * 2 + pw;

  const int wid  = threadIdx.x >> 5;
  const int lane = threadIdx.x & 31;
  const int numNT = (inS + 15) >> 4;
  const bool valid = (blockIdx.x * 8 + wid) < numNT;
  int nTile = blockIdx.x * 8 + wid;
  if (nTile >= numNT) nTile = numNT - 1;
  const int nBase  = nTile << 4;
  const int coBase = blockIdx.y << 4;

  __shared__ __bf16 lA[8][16 * 32];
  __shared__ __bf16 lB[8][32 * 16];
  __bf16* sA = lA[wid];
  __bf16* sB = lB[wid];

  v8f acc = {};
  const int n     = lane & 15;
  const int khalf = (lane & 16) ? 8 : 0;

  for (int kb = 0; kb < Ci; kb += 32) {
    for (int t = lane; t < 512; t += 32) {
      int m = t >> 5, kk = t & 31;
      int co = coBase + m, ci = kb + kk;
      float v = 0.f;
      if (co < Co && ci < Ci) v = w[((size_t)ci * Co + co) * 8 + widx];
      sA[t] = (__bf16)v;
    }
    for (int t = lane; t < 512; t += 32) {
      int kk = t >> 4, nn = t & 15;
      int ci = kb + kk;
      int voxel = nBase + nn;
      float v = 0.f;
      if (ci < Ci && voxel < inS) v = x[(size_t)ci * inS + voxel];
      sB[t] = (__bf16)v;
    }
    __syncthreads();

    Frag16 a, b;
    const unsigned short* sA16 = (const unsigned short*)sA;
    const unsigned short* sB16 = (const unsigned short*)sB;
    const int m = lane & 15;
#pragma unroll
    for (int j = 0; j < 8; ++j) {
      int k0 = (j < 4 ? 2 * j : 16 + 2 * (j - 4)) + khalf;
      a.u[j] = *(const unsigned int*)(sA16 + m * 32 + k0);
      unsigned lo = sB16[k0 * 16 + n];
      unsigned hi = sB16[(k0 + 1) * 16 + n];
      b.u[j] = lo | (hi << 16);
    }
    acc = __builtin_amdgcn_wmma_f32_16x16x32_bf16(
        false, a.v, false, b.v, (short)0, acc, false, false);
    __syncthreads();
  }

  const int rowOff = (lane & 16) ? 8 : 0;
  const int voxel = nBase + n;
  if (valid && voxel < inS) {
    int vd = voxel / inHW; int r2 = voxel - vd * inHW;
    int vh = r2 / inW;     int vw = r2 - vh * inW;
    int od = 2 * vd + pd, oh = 2 * vh + ph, ow = 2 * vw + pw;
    size_t obase = (size_t)od * outHW + (size_t)oh * outW + ow;
#pragma unroll
    for (int r = 0; r < 8; ++r) {
      int co = coBase + r + rowOff;
      if (co < Co) y[(size_t)co * outS + obase] = acc[r];
    }
  }
}

// ---------------------------------------------------------------------------
// Host orchestration of the full U-Net forward pass.
// ---------------------------------------------------------------------------
extern "C" void kernel_launch(void* const* d_in, const int* in_sizes, int n_in,
                              void* d_out, int out_size, void* d_ws, size_t ws_size,
                              hipStream_t stream) {
  (void)in_sizes; (void)n_in; (void)out_size; (void)ws_size;
  auto P = [&](int i) { return (const float*)d_in[i]; };

  const int S96 = 96 * 96 * 96, S48 = 48 * 48 * 48, S24 = 24 * 24 * 24;
  const int S12 = 12 * 12 * 12, S6 = 216, S3 = 27;

  float* base = (float*)d_ws;
  size_t sp = 0;
  auto alloc = [&](size_t nfl) { float* p = base + sp; sp += (nfl + 63) & ~(size_t)63; return p; };

  float* stats = alloc(1024);
  float* mean = stats;
  float* inv  = stats + 512;
  const size_t TMP = (size_t)16 * S48;        // largest block-temp (16ch @ 48^3)
  float* tmpA = alloc(TMP);
  float* tmpB = alloc(TMP);
  // skip-connection tensors (pre-BN conv outputs, reused by decoder concats)
  float* e2 = alloc((size_t)32 * S24);
  float* e3 = alloc((size_t)64 * S12);
  float* e4 = alloc((size_t)128 * S6);
  float* e5 = alloc((size_t)256 * S3);

  auto conv = [&](const float* xA, int CiA, const float* xB, const float* w,
                  float* y, int Ci, int Co, int inN, int outN, int k) {
    int outS = outN * outN * outN;
    int numNT = (outS + 31) / 32;
    dim3 g((numNT + 7) / 8, (Co + 15) / 16, 1);
    if (k == 3)
      k_conv_wmma<3, 1, 1><<<g, 256, 0, stream>>>(xA, CiA, xB, w, y, Ci, Co,
                                                  inN, inN, inN, outN, outN, outN);
    else if (k == 2)
      k_conv_wmma<2, 2, 0><<<g, 256, 0, stream>>>(xA, CiA, xB, w, y, Ci, Co,
                                                  inN, inN, inN, outN, outN, outN);
    else
      k_conv_wmma<1, 1, 0><<<g, 256, 0, stream>>>(xA, CiA, xB, w, y, Ci, Co,
                                                  inN, inN, inN, outN, outN, outN);
  };
  auto tconv = [&](const float* x, const float* w, float* y, int Ci, int Co, int inN) {
    int inS = inN * inN * inN;
    int numNT = (inS + 15) / 16;
    dim3 g((numNT + 7) / 8, (Co + 15) / 16, 8);
    k_tconv_wmma<<<g, 256, 0, stream>>>(x, w, y, Ci, Co, inN, inN, inN);
  };
  auto affine = [&](const float* x, const float* res, float* y,
                    const float* g, const float* b, int C, int S, int flags) {
    long tot = (long)C * S;
    k_affine<<<(unsigned)((tot + 255) / 256), 256, 0, stream>>>(
        x, res, y, mean, inv, g, b, C, S, flags);
  };
  auto bnrelu = [&](const float* x, float* y, const float* g, const float* b, int C, int S) {
    k_bnstats<<<C, 256, 0, stream>>>(x, S, mean, inv);
    affine(x, nullptr, y, g, b, C, S, FLAG_BN | FLAG_RELU);
  };
  // BasicBlock: relu(x + bn(conv3(relu(bn(conv3(x))))))
  // param layout at cbase: c1, n1_g, n1_b, c2, n2_g, n2_b
  auto bb = [&](const float* x, float* out, int C, int N, int cbase) {
    int S = N * N * N;
    conv(x, C, nullptr, P(cbase + 0), tmpA, C, C, N, N, 3);
    k_bnstats<<<C, 256, 0, stream>>>(tmpA, S, mean, inv);
    affine(tmpA, nullptr, tmpA, P(cbase + 1), P(cbase + 2), C, S, FLAG_BN | FLAG_RELU);
    conv(tmpA, C, nullptr, P(cbase + 3), tmpB, C, C, N, N, 3);
    k_bnstats<<<C, 256, 0, stream>>>(tmpB, S, mean, inv);
    affine(tmpB, x, out, P(cbase + 4), P(cbase + 5), C, S, FLAG_BN | FLAG_RES | FLAG_RELU);
  };
  auto cat2 = [&](float* dst, const float* a, const float* b2, size_t nflA, size_t nflB) {
    hipMemcpyAsync(dst, a, nflA * sizeof(float), hipMemcpyDeviceToDevice, stream);
    hipMemcpyAsync(dst + nflA, b2, nflB * sizeof(float), hipMemcpyDeviceToDevice, stream);
  };

  // ------------------ encoder ------------------
  size_t mark0 = sp;
  float* e1   = alloc((size_t)16 * S48);
  float* e1a  = alloc((size_t)16 * S48);
  float* e1_1 = alloc((size_t)16 * S48);
  size_t markE1 = sp;
  float* e0_1 = alloc((size_t)8 * S96);

  // eb0: 1x1 conv on [x_geo|x_col] (dual-input, concat never materialized) + relu
  conv(P(0), 1, P(1), P(2), e0_1, 4, 8, 96, 96, 1);
  affine(e0_1, nullptr, e0_1, nullptr, nullptr, 8, S96, FLAG_RELU);
  conv(e0_1, 8, nullptr, P(3), e1, 8, 16, 96, 48, 2);                // ec0
  sp = markE1;                                                        // free e0_1

  bnrelu(e1, e1a, P(8), P(9), 16, S48);
  bb(e1a, e1_1, 16, 48, 10);
  conv(e1_1, 16, nullptr, P(4), e2, 16, 32, 48, 24, 2);              // ec1
  sp = mark0;

  { float* a = alloc((size_t)32 * S24); float* o = alloc((size_t)32 * S24);
    bnrelu(e2, a, P(16), P(17), 32, S24);
    bb(a, o, 32, 24, 18);
    conv(o, 32, nullptr, P(5), e3, 32, 64, 24, 12, 2); }             // ec2
  sp = mark0;

  { float* a = alloc((size_t)64 * S12); float* o = alloc((size_t)64 * S12);
    bnrelu(e3, a, P(24), P(25), 64, S12);
    bb(a, o, 64, 12, 26);
    conv(o, 64, nullptr, P(6), e4, 64, 128, 12, 6, 2); }             // ec3
  sp = mark0;

  { float* a = alloc((size_t)128 * S6); float* o = alloc((size_t)128 * S6);
    bnrelu(e4, a, P(32), P(33), 128, S6);
    bb(a, o, 128, 6, 34);
    conv(o, 128, nullptr, P(7), e5, 128, 256, 6, 3, 2); }            // ec4
  sp = mark0;

  // ------------------ bottleneck ------------------
  float* e5a   = alloc((size_t)256 * S3);
  bnrelu(e5, e5a, P(40), P(41), 256, S3);
  float* bneck = alloc((size_t)256 * S3);
  bb(e5a, bneck, 256, 3, 42);

  // ------------------ decoder ------------------
  // dec4 @ 3^3, 512ch   (params: db4 base 48)
  float* c45  = alloc((size_t)512 * S3); cat2(c45, bneck, e5, (size_t)256 * S3, (size_t)256 * S3);
  float* bb4  = alloc((size_t)512 * S3); bb(c45, bb4, 512, 3, 48);
  float* t4   = alloc((size_t)512 * S3); conv(bb4, 512, nullptr, P(54), t4, 512, 512, 3, 3, 3);
  float* de4f = alloc((size_t)512 * S3); bnrelu(t4, de4f, P(55), P(56), 512, S3);

  float* d3 = alloc((size_t)128 * S6);   tconv(de4f, P(84), d3, 512, 128, 3);   // du4

  // dec3 @ 6^3, 256ch   (params: db3 base 57)
  float* c3   = alloc((size_t)256 * S6); cat2(c3, d3, e4, (size_t)128 * S6, (size_t)128 * S6);
  float* bb3  = alloc((size_t)256 * S6); bb(c3, bb3, 256, 6, 57);
  float* t3   = alloc((size_t)256 * S6); conv(bb3, 256, nullptr, P(63), t3, 256, 256, 6, 6, 3);
  float* de3f = alloc((size_t)256 * S6); bnrelu(t3, de3f, P(64), P(65), 256, S6);

  float* d2 = alloc((size_t)64 * S12);   tconv(de3f, P(85), d2, 256, 64, 6);    // du3

  // dec2 @ 12^3, 128ch  (params: db2 base 66)
  float* c2b  = alloc((size_t)128 * S12); cat2(c2b, d2, e3, (size_t)64 * S12, (size_t)64 * S12);
  float* bb2  = alloc((size_t)128 * S12); bb(c2b, bb2, 128, 12, 66);
  float* t2   = alloc((size_t)128 * S12); conv(bb2, 128, nullptr, P(72), t2, 128, 128, 12, 12, 3);
  float* de2f = alloc((size_t)128 * S12); bnrelu(t2, de2f, P(73), P(74), 128, S12);

  float* d1 = alloc((size_t)32 * S24);    tconv(de2f, P(86), d1, 128, 32, 12);  // du2

  // rematch + dec1 @ 24^3, 64ch  (params: db1 base 75)
  float* de_1 = alloc((size_t)64 * S24);  cat2(de_1, d1, e2, (size_t)32 * S24, (size_t)32 * S24);
  float* rem  = alloc((size_t)64 * S24);  conv(de_1, 64, nullptr, P(87), rem, 64, 64, 24, 24, 3);
  float* bb1  = alloc((size_t)64 * S24);  bb(rem, bb1, 64, 24, 75);
  float* t1   = alloc((size_t)64 * S24);  conv(bb1, 64, nullptr, P(81), t1, 64, 64, 24, 24, 3);
  bnrelu(t1, (float*)d_out, P(82), P(83), 64, S24);
}
